// MultiHeadAttn_19928648253502
// MI455X (gfx1250) — compile-verified, matrix operand order checked by reading, not run on hardware
//
#include <hip/hip_runtime.h>
#include <hip/hip_bf16.h>
#include <stdint.h>

// ---------------------------------------------------------------------------
// MI455X (gfx1250) MHA layer: bf16 WMMA GEMMs + flash attention + layernorm.
// Wave32; v_wmma_f32_16x16x32_bf16 for all matmuls; global_load_async_to_lds
// (ASYNCcnt) double-buffered tile staging in the GEMMs.
// ---------------------------------------------------------------------------

typedef __attribute__((ext_vector_type(16))) __bf16 v16bf;
typedef __attribute__((ext_vector_type(8)))  __bf16 v8bf;
typedef __attribute__((ext_vector_type(8)))  float  v8f;

#define B_  4
#define S_  2048
#define D_  1024
#define H_  16
#define DK_ 64
#define DV_ 64

union Frag { v16bf v; v8bf p[2]; };

__device__ __forceinline__ __bf16 f2bf(float f) {
  union { float f; uint32_t u; } x; x.f = f;
  uint32_t r = x.u + 0x7FFFu + ((x.u >> 16) & 1u);   // round-to-nearest-even
  uint16_t hh = (uint16_t)(r >> 16);
  __bf16 o;
  __builtin_memcpy(&o, &hh, sizeof(o));
  return o;
}

// CDNA5 async global->LDS copy, 16 bytes per lane. VDST operand is the LDS
// byte address in a VGPR (low 32 bits of the flat shared pointer).
__device__ __forceinline__ void async_b128(void* lds, const void* gptr) {
  uint32_t lo = (uint32_t)(uintptr_t)lds;
  uint64_t ga = (uint64_t)(uintptr_t)gptr;
  asm volatile("global_load_async_to_lds_b128 %0, %1, off"
               :: "v"(lo), "v"(ga) : "memory");
}
__device__ __forceinline__ void wait_async() {
  asm volatile("s_wait_asynccnt 0" ::: "memory");
}

// --------------------------- f32 -> bf16 pack ------------------------------
__global__ void cast_f32_bf16(const float* __restrict__ in,
                              __bf16* __restrict__ out, int n) {
  int i = blockIdx.x * blockDim.x + threadIdx.x;
  int stride = gridDim.x * blockDim.x;
  for (; i < n; i += stride) out[i] = f2bf(in[i]);
}

// --------------------------- WMMA GEMM -------------------------------------
// C[M,N] = A[M,K] * Bt[N,K]^T   (A row-major bf16, Bt row-major bf16)
// Tile: 128x128, BK=32. 256 threads = 8 waves laid out 4(M) x 2(N),
// each wave computes 32x64 = 2x4 WMMA tiles. LDS tiles are double-buffered
// and filled with global_load_async_to_lds_b128 (ASYNCcnt), so the copy of
// tile k+1 overlaps WMMA on tile k with one barrier per K-step.
template <bool OUT_BF16>
__global__ __launch_bounds__(256)
void gemm_bf16_wmma(const __bf16* __restrict__ A,
                    const __bf16* __restrict__ Bt,
                    void* __restrict__ Cout,
                    int M, int N, int K) {
  constexpr int BM = 128, BN = 128, BK = 32;
  __shared__ alignas(16) __bf16 As[2][BM][BK + 8];
  __shared__ alignas(16) __bf16 Bs[2][BN][BK + 8];

  const int tid   = threadIdx.x;
  const int lane  = tid & 31;
  const int wave  = tid >> 5;
  const int wm    = (wave & 3) * 32;      // wave row offset
  const int wn    = (wave >> 2) * 64;     // wave col offset
  const int mlo   = lane & 15;
  const int khalf = (lane >> 4) << 3;     // A/B fragment K split (0 or 8)
  const int rhalf = (lane >> 4) << 3;     // C fragment row split

  const int m0 = blockIdx.x * BM;
  const int n0 = blockIdx.y * BN;

  const int ldRow = tid >> 1;             // 0..127
  const int ldCol = (tid & 1) << 4;       // 0 or 16

  v8f acc[2][4] = {};

  // issue async copy of one K-slab into buffer `buf`
  auto issue_tile = [&](int buf, int k0) {
    const __bf16* ag = A + (size_t)(m0 + ldRow) * K + k0 + ldCol;
    async_b128(&As[buf][ldRow][ldCol],     ag);
    async_b128(&As[buf][ldRow][ldCol + 8], ag + 8);
    const __bf16* bg = Bt + (size_t)(n0 + ldRow) * K + k0 + ldCol;
    async_b128(&Bs[buf][ldRow][ldCol],     bg);
    async_b128(&Bs[buf][ldRow][ldCol + 8], bg + 8);
  };

  issue_tile(0, 0);

  for (int k0 = 0, it = 0; k0 < K; k0 += BK, ++it) {
    const int cur = it & 1;
    wait_async();
    __syncthreads();                      // buf `cur` written; buf `cur^1` free
    if (k0 + BK < K) issue_tile(cur ^ 1, k0 + BK);

    Frag bf[4];
#pragma unroll
    for (int nt = 0; nt < 4; ++nt) {
      const int c = wn + nt * 16 + mlo;
      bf[nt].p[0] = *(const v8bf*)&Bs[cur][c][khalf];
      bf[nt].p[1] = *(const v8bf*)&Bs[cur][c][16 + khalf];
    }
#pragma unroll
    for (int mt = 0; mt < 2; ++mt) {
      Frag a;
      const int r = wm + mt * 16 + mlo;
      a.p[0] = *(const v8bf*)&As[cur][r][khalf];
      a.p[1] = *(const v8bf*)&As[cur][r][16 + khalf];
#pragma unroll
      for (int nt = 0; nt < 4; ++nt)
        acc[mt][nt] = __builtin_amdgcn_wmma_f32_16x16x32_bf16(
            false, a.v, false, bf[nt].v, (short)0, acc[mt][nt], false, false);
    }
  }

#pragma unroll
  for (int mt = 0; mt < 2; ++mt)
#pragma unroll
    for (int nt = 0; nt < 4; ++nt)
#pragma unroll
      for (int r = 0; r < 8; ++r) {
        const int row = m0 + wm + mt * 16 + r + rhalf;
        const int col = n0 + wn + nt * 16 + mlo;
        if (OUT_BF16)
          ((__bf16*)Cout)[(size_t)row * N + col] = f2bf(acc[mt][nt][r]);
        else
          ((float*)Cout)[(size_t)row * N + col] = acc[mt][nt][r];
      }
}

// --------------------------- Flash attention -------------------------------
// One workgroup (4 waves / 128 threads) per (b, h, 64-row query block).
// Each wave owns 16 query rows; online softmax over 32 key blocks of 64.
// Q/K tiles staged with async global->LDS; V staged through VGPRs (needs a
// transpose on the LDS store to form the B^T layout for the PV matmul).
__global__ __launch_bounds__(128)
void flash_attn(const __bf16* __restrict__ hq,
                const __bf16* __restrict__ hk,
                const __bf16* __restrict__ hv,
                __bf16* __restrict__ heads) {
  __shared__ alignas(16) __bf16 Qs[64][72];
  __shared__ alignas(16) __bf16 Kt[64][72];       // [key][dk]  (= B^T for QK^T)
  __shared__ alignas(16) __bf16 Vs[64][72];       // [dv][key]  (= B^T for PV)
  __shared__ alignas(16) __bf16 Ps[4][16][72];    // per-wave P tile (C->A relayout)

  const int tid   = threadIdx.x;
  const int lane  = tid & 31;
  const int w     = tid >> 5;
  const int mlo   = lane & 15;
  const int khalf = (lane >> 4) << 3;
  const int rhalf = (lane >> 4) << 3;

  const int q0 = blockIdx.x * 64;
  const int h  = blockIdx.y;
  const int b  = blockIdx.z;

  const size_t rowStride = (size_t)H_ * DK_;                  // 1024
  const size_t base = ((size_t)b * S_) * rowStride + (size_t)h * DK_;

  const int r  = tid >> 1;                 // 0..63
  const int c  = (tid & 1) << 5;           // 0 or 32

  {   // stage Q block 64x64 (async)
    const __bf16* src = hq + base + (size_t)(q0 + r) * rowStride + c;
#pragma unroll
    for (int i = 0; i < 4; ++i)
      async_b128(&Qs[r][c + i * 8], src + i * 8);
    wait_async();
  }
  __syncthreads();

  Frag qa[2];
#pragma unroll
  for (int kt = 0; kt < 2; ++kt) {
    qa[kt].p[0] = *(const v8bf*)&Qs[w * 16 + mlo][kt * 32 + khalf];
    qa[kt].p[1] = *(const v8bf*)&Qs[w * 16 + mlo][kt * 32 + 16 + khalf];
  }

  v8f oacc[4] = {};
  float mrow[8], lrow[8];
#pragma unroll
  for (int rr = 0; rr < 8; ++rr) { mrow[rr] = -3.0e38f; lrow[rr] = 0.f; }

  const float invT = 0.03125f;   // 1/sqrt(D)=1/32 (ref scales by sqrt(d_model))

  for (int j = 0; j < S_ / 64; ++j) {
    const int kb0 = j * 64;
    {   // K row-major via async; V transposed via VGPR path
      const __bf16* ksrc = hk + base + (size_t)(kb0 + r) * rowStride + c;
#pragma unroll
      for (int i = 0; i < 4; ++i)
        async_b128(&Kt[r][c + i * 8], ksrc + i * 8);
      const __bf16* vsrc = hv + base + (size_t)(kb0 + r) * rowStride + c;
#pragma unroll
      for (int i = 0; i < 4; ++i) {
        v8bf vv = *(const v8bf*)(vsrc + i * 8);
#pragma unroll
        for (int e = 0; e < 8; ++e) Vs[c + i * 8 + e][r] = vv[e];
      }
      wait_async();
    }
    __syncthreads();

    // S = Q * K^T  (16x64 per wave)
    v8f sacc[4] = {};
#pragma unroll
    for (int kt = 0; kt < 2; ++kt)
#pragma unroll
      for (int nt = 0; nt < 4; ++nt) {
        Frag kf;
        kf.p[0] = *(const v8bf*)&Kt[nt * 16 + mlo][kt * 32 + khalf];
        kf.p[1] = *(const v8bf*)&Kt[nt * 16 + mlo][kt * 32 + 16 + khalf];
        sacc[nt] = __builtin_amdgcn_wmma_f32_16x16x32_bf16(
            false, qa[kt].v, false, kf.v, (short)0, sacc[nt], false, false);
      }

    // online softmax (row stats across the 16-lane halves of the C layout)
#pragma unroll
    for (int rr = 0; rr < 8; ++rr) {
      float vmax = -3.0e38f;
#pragma unroll
      for (int nt = 0; nt < 4; ++nt) {
        sacc[nt][rr] *= invT;
        vmax = fmaxf(vmax, sacc[nt][rr]);
      }
#pragma unroll
      for (int off = 1; off < 16; off <<= 1)
        vmax = fmaxf(vmax, __shfl_xor(vmax, off, 32));
      const float mnew = fmaxf(mrow[rr], vmax);
      const float corr = __expf(mrow[rr] - mnew);
      mrow[rr] = mnew;
      float rsum = 0.f;
#pragma unroll
      for (int nt = 0; nt < 4; ++nt) {
        const float p = __expf(sacc[nt][rr] - mnew);
        rsum += p;
        Ps[w][rr + rhalf][nt * 16 + mlo] = f2bf(p);
      }
#pragma unroll
      for (int off = 1; off < 16; off <<= 1)
        rsum += __shfl_xor(rsum, off, 32);
      lrow[rr] = lrow[rr] * corr + rsum;
#pragma unroll
      for (int nt = 0; nt < 4; ++nt) oacc[nt][rr] *= corr;
    }

    // O += P * V  (P re-read from LDS in A-fragment layout)
#pragma unroll
    for (int kt = 0; kt < 2; ++kt) {
      Frag pa;
      pa.p[0] = *(const v8bf*)&Ps[w][mlo][kt * 32 + khalf];
      pa.p[1] = *(const v8bf*)&Ps[w][mlo][kt * 32 + 16 + khalf];
#pragma unroll
      for (int nt = 0; nt < 4; ++nt) {
        Frag vf;
        vf.p[0] = *(const v8bf*)&Vs[nt * 16 + mlo][kt * 32 + khalf];
        vf.p[1] = *(const v8bf*)&Vs[nt * 16 + mlo][kt * 32 + 16 + khalf];
        oacc[nt] = __builtin_amdgcn_wmma_f32_16x16x32_bf16(
            false, pa.v, false, vf.v, (short)0, oacc[nt], false, false);
      }
    }
    __syncthreads();
  }

  // write concat layout [B*S, H*DV] in bf16 (feeds output-projection GEMM)
#pragma unroll
  for (int nt = 0; nt < 4; ++nt)
#pragma unroll
    for (int rr = 0; rr < 8; ++rr) {
      const int row = q0 + w * 16 + rr + rhalf;
      const int col = h * DV_ + nt * 16 + mlo;
      heads[((size_t)b * S_ + row) * (H_ * DV_) + col] =
          f2bf(oacc[nt][rr] / lrow[rr]);
    }
}

// --------------------------- residual + LayerNorm --------------------------
__global__ __launch_bounds__(256)
void layernorm_res(const float* __restrict__ proj, const float* __restrict__ resid,
                   const float* __restrict__ scale, const float* __restrict__ offset,
                   float* __restrict__ out) {
  const int row = blockIdx.x;
  const int tid = threadIdx.x;
  const size_t basei = (size_t)row * D_;
  float x[4], s = 0.f, ss = 0.f;
#pragma unroll
  for (int j = 0; j < 4; ++j) {
    const int i = tid + j * 256;
    x[j] = proj[basei + i] + resid[basei + i];
    s += x[j];
    ss += x[j] * x[j];
  }
#pragma unroll
  for (int off = 1; off < 32; off <<= 1) {
    s  += __shfl_xor(s, off, 32);
    ss += __shfl_xor(ss, off, 32);
  }
  __shared__ float red[2][8];
  const int w = tid >> 5;
  if ((tid & 31) == 0) { red[0][w] = s; red[1][w] = ss; }
  __syncthreads();
  if (tid < 32) {
    float a  = (tid < 8) ? red[0][tid] : 0.f;
    float bq = (tid < 8) ? red[1][tid] : 0.f;
#pragma unroll
    for (int off = 1; off < 8; off <<= 1) {
      a  += __shfl_xor(a, off, 32);
      bq += __shfl_xor(bq, off, 32);
    }
    if (tid == 0) { red[0][0] = a; red[1][0] = bq; }
  }
  __syncthreads();
  const float mean = red[0][0] * (1.0f / D_);
  const float var  = fmaxf((red[1][0] - (float)D_ * mean * mean) * (1.0f / (D_ - 1)), 0.f);
  const float inv  = 1.0f / (sqrtf(var) + 1e-9f);
#pragma unroll
  for (int j = 0; j < 4; ++j) {
    const int i = tid + j * 256;
    out[basei + i] = scale[i] * (x[j] - mean) * inv + offset[i];
  }
}

// --------------------------- driver ----------------------------------------
extern "C" void kernel_launch(void* const* d_in, const int* in_sizes, int n_in,
                              void* d_out, int out_size, void* d_ws, size_t ws_size,
                              hipStream_t stream) {
  (void)in_sizes; (void)n_in; (void)out_size; (void)ws_size;
  const float* q      = (const float*)d_in[0];
  const float* k      = (const float*)d_in[1];
  const float* v      = (const float*)d_in[2];
  const float* Wq     = (const float*)d_in[3];
  const float* Wk     = (const float*)d_in[4];
  const float* Wv     = (const float*)d_in[5];
  const float* Wp     = (const float*)d_in[6];
  const float* scale  = (const float*)d_in[7];
  const float* offset = (const float*)d_in[8];
  float* out = (float*)d_out;

  const size_t ACT = (size_t)B_ * S_ * D_;     // 8,388,608 elems
  const size_t WEL = (size_t)H_ * DK_ * D_;    // 1,048,576 elems

  char* p = (char*)d_ws;
  __bf16* qb  = (__bf16*)p; p += ACT * 2;
  __bf16* kb  = (__bf16*)p; p += ACT * 2;
  __bf16* vb  = (__bf16*)p; p += ACT * 2;
  __bf16* Wqb = (__bf16*)p; p += WEL * 2;
  __bf16* Wkb = (__bf16*)p; p += WEL * 2;
  __bf16* Wvb = (__bf16*)p; p += WEL * 2;
  __bf16* Wpb = (__bf16*)p; p += WEL * 2;
  __bf16* hq  = (__bf16*)p; p += ACT * 2;
  __bf16* hk  = (__bf16*)p; p += ACT * 2;
  __bf16* hv  = (__bf16*)p; p += ACT * 2;
  __bf16* hds = (__bf16*)p; p += ACT * 2;
  float*  prj = (float*)p;                     // ACT * 4 bytes

  // 1) pack activations + weights to bf16
  cast_f32_bf16<<<8192, 256, 0, stream>>>(q,  qb,  (int)ACT);
  cast_f32_bf16<<<8192, 256, 0, stream>>>(k,  kb,  (int)ACT);
  cast_f32_bf16<<<8192, 256, 0, stream>>>(v,  vb,  (int)ACT);
  cast_f32_bf16<<<1024, 256, 0, stream>>>(Wq, Wqb, (int)WEL);
  cast_f32_bf16<<<1024, 256, 0, stream>>>(Wk, Wkb, (int)WEL);
  cast_f32_bf16<<<1024, 256, 0, stream>>>(Wv, Wvb, (int)WEL);
  cast_f32_bf16<<<1024, 256, 0, stream>>>(Wp, Wpb, (int)WEL);

  // 2) head projections: [8192,1024] x [1024,1024]^T   (weights already N x K)
  const int M = B_ * S_, N = H_ * DK_, K = D_;
  dim3 gg(M / 128, N / 128);
  gemm_bf16_wmma<true><<<gg, 256, 0, stream>>>(qb, Wqb, hq, M, N, K);
  gemm_bf16_wmma<true><<<gg, 256, 0, stream>>>(kb, Wkb, hk, M, N, K);
  gemm_bf16_wmma<true><<<gg, 256, 0, stream>>>(vb, Wvb, hv, M, N, K);

  // 3) flash attention -> concat heads [8192, 1024] bf16
  flash_attn<<<dim3(S_ / 64, H_, B_), 128, 0, stream>>>(hq, hk, hv, hds);

  // 4) output projection: [8192,1024] x Wp[1024,1024]^T -> f32
  gemm_bf16_wmma<false><<<gg, 256, 0, stream>>>(hds, Wpb, prj, M, D_, H_ * DV_);

  // 5) residual + layernorm (ddof=1, std+eps)
  layernorm_res<<<M, 256, 0, stream>>>(prj, q, scale, offset, out);
}
